// Informer_41858751267394
// MI455X (gfx1250) — matrix operand order repaired and checked by reading, not compiled, not run on hardware
//
#include <hip/hip_runtime.h>
#include <cstdint>

// ---------------------------------------------------------------------------
// Informer forward for MI455X (gfx1250, wave32, WMMA).
// GEMM-class work -> v_wmma_f32_16x16x32_bf16, f32 accumulate, 4 accumulators
// per wave (16x64 tile) so one A fragment feeds 4 WMMAs. The A macro-tile is
// staged into LDS with global_load_async_to_lds_b128 (ASYNCcnt +
// s_wait_asynccnt), since it is shared by the 4 N-waves of the block.
// Bias / GELU / residual fused into epilogues (forward is HBM-bound at
// 23.3 TB/s; compute is far below the WMMA ceiling).
// ---------------------------------------------------------------------------

static constexpr int DM  = 512;   // d_model
static constexpr int NH  = 8;     // heads
static constexpr int HD  = 64;    // head dim
static constexpr int DFF = 2048;  // ffn dim
static constexpr int BB  = 8;     // batch
static constexpr int SL  = 1024;  // encoder seq len
static constexpr int DLN = 512;   // decoder seq len (label+pred)

typedef __attribute__((ext_vector_type(16))) __bf16 v16bf;
typedef __attribute__((ext_vector_type(8)))  float  v8f;

__device__ __forceinline__ __bf16 f2bf(float f) { return (__bf16)f; }

__device__ __forceinline__ v8f wmma_bf16(const v16bf& a, const v16bf& b, const v8f& c) {
  return __builtin_amdgcn_wmma_f32_16x16x32_bf16(false, a, false, b,
                                                 (short)0, c, false, false);
}

// ---------------------------------------------------------------------------
// C[M,N] = act(A[M,K] @ W[N,K]^T + bias) (+ residual).
// Block: 8 waves = 2(M) x 4(N) -> 32 x 256 macro tile; each wave: 16x64 via
// 4 accumulators. K%32==0, M%32==0, N%256==0.
// A macro tile (32x32 f32, 4KB) is async-copied to LDS each K step:
// 256 threads x one b128 async copy, then s_wait_asynccnt + barrier.
// Fragment packing per CDNA5 ISA 7.12.2:
//   A (16x32 bf16): lane half 0 -> K {k..k+7, k+16..k+23}, half 1 -> +8
//   B (32x16 bf16): same K packing as A
//   C (16x16 f32) : lane = N col, vgpr i = M row (lanes 16-31: row+8)
// ---------------------------------------------------------------------------
template <int ACT, bool RES>
__global__ __launch_bounds__(256) void gemm_bias_act(
    const float* __restrict__ A, int lda, const float* __restrict__ W, int ldw,
    const float* __restrict__ bias, const float* __restrict__ R, int ldr,
    float* __restrict__ C, int ldc, int K) {
  __shared__ float As[32 * 32];
  const int tid  = threadIdx.x;
  const int lane = tid & 31;
  const int wave = tid >> 5;
  const int row  = lane & 15;
  const int half = lane >> 4;
  const int tm0  = blockIdx.y * 32;
  const int tm   = tm0 + (wave >> 2) * 16;
  const int tn   = blockIdx.x * 256 + (wave & 3) * 64;

  // async stage mapping: thread -> (row, 4-float chunk) of the 32x32 A tile
  const int sr  = tid >> 3;
  const int sc4 = (tid & 7) * 4;
  const float*   ag      = A + (size_t)(tm0 + sr) * lda + sc4;
  const unsigned lds_dst = (unsigned)(uintptr_t)(&As[sr * 32 + sc4]);

  const float* wp[4];
#pragma unroll
  for (int s = 0; s < 4; ++s) wp[s] = W + (size_t)(tn + s * 16 + row) * ldw;
  const float* arow = &As[((wave >> 2) * 16 + row) * 32];

  v8f acc[4] = {v8f{}, v8f{}, v8f{}, v8f{}};
  for (int k = 0; k < K; k += 32) {
    __syncthreads();  // previous-iteration readers done before overwrite
    asm volatile("global_load_async_to_lds_b128 %0, %1, off"
                 :: "v"(lds_dst), "v"(ag + k) : "memory");
#pragma unroll
    for (int s = 0; s < 4; ++s) __builtin_prefetch(wp[s] + k + 128, 0, 1);
    asm volatile("s_wait_asynccnt 0" ::: "memory");
    __syncthreads();  // all waves' async stages visible

    const int ka = half * 8;        // LDS-local K offset
    const int kb = k + half * 8;    // global K offset for B
    v16bf av;
#pragma unroll
    for (int i = 0; i < 8; ++i) {
      av[i]     = f2bf(arow[ka + i]);
      av[i + 8] = f2bf(arow[ka + 16 + i]);
    }
#pragma unroll
    for (int s = 0; s < 4; ++s) {
      v16bf bv;
#pragma unroll
      for (int j = 0; j < 16; ++j) bv[j] = f2bf(wp[s][kb + (j < 8 ? j : j + 8)]);
      acc[s] = wmma_bf16(av, bv, acc[s]);
    }
  }
#pragma unroll
  for (int s = 0; s < 4; ++s) {
    const int col = tn + s * 16 + row;
    const float bc = bias ? bias[col] : 0.0f;
#pragma unroll
    for (int i = 0; i < 8; ++i) {
      const int m = tm + half * 8 + i;
      float v = acc[s][i] + bc;
      if (ACT == 1) v = 0.5f * v * (1.0f + erff(v * 0.70710678118f));  // gelu
      if (RES) v += R[(size_t)m * ldr + col];
      C[(size_t)m * ldc + col] = v;
    }
  }
}

// ---------------------------------------------------------------------------
// S[h,l,s] = scale * Q[l,h,:] . K[s,h,:] (per batch; z = head), causal opt.
// Wave computes 16x64; block 32x256 of the score matrix.
// ---------------------------------------------------------------------------
__global__ __launch_bounds__(256) void attn_scores(
    const float* __restrict__ Q, const float* __restrict__ Km,
    float* __restrict__ S, int Lq, int Lk, float scale, int causal) {
  const int lane = threadIdx.x & 31;
  const int wave = threadIdx.x >> 5;
  const int row  = lane & 15;
  const int half = lane >> 4;
  const int h  = blockIdx.z;
  const int tm = blockIdx.y * 32 + (wave >> 2) * 16;
  const int tn = blockIdx.x * 256 + (wave & 3) * 64;
  const float* q = Q + (size_t)(tm + row) * DM + h * HD;
  const float* kp[4];
#pragma unroll
  for (int s = 0; s < 4; ++s) kp[s] = Km + (size_t)(tn + s * 16 + row) * DM + h * HD;
  v8f acc[4] = {v8f{}, v8f{}, v8f{}, v8f{}};
#pragma unroll
  for (int k = 0; k < HD; k += 32) {
    const int ka = k + half * 8;
    const int kb = k + half * 8;
    v16bf av;
#pragma unroll
    for (int i = 0; i < 8; ++i) {
      av[i]     = f2bf(q[ka + i]);
      av[i + 8] = f2bf(q[ka + 16 + i]);
    }
#pragma unroll
    for (int s = 0; s < 4; ++s) {
      v16bf bv;
#pragma unroll
      for (int j = 0; j < 16; ++j) bv[j] = f2bf(kp[s][kb + (j < 8 ? j : j + 8)]);
      acc[s] = wmma_bf16(av, bv, acc[s]);
    }
  }
  float* sp = S + (size_t)h * Lq * Lk;
#pragma unroll
  for (int s = 0; s < 4; ++s) {
    const int col = tn + s * 16 + row;
#pragma unroll
    for (int i = 0; i < 8; ++i) {
      const int m = tm + half * 8 + i;
      float v = acc[s][i] * scale;
      if (causal && col > m) v = -__builtin_inff();
      sp[(size_t)m * Lk + col] = v;
    }
  }
}

// O[l, h*64+d] = sum_s P[h,l,s] * V[s, h*64+d]; wave covers a full head (N=64)
__global__ __launch_bounds__(256) void attn_av(
    const float* __restrict__ P, const float* __restrict__ V,
    float* __restrict__ O, int Lq, int Lk) {
  const int lane = threadIdx.x & 31;
  const int wave = threadIdx.x >> 5;
  const int row  = lane & 15;
  const int half = lane >> 4;
  const int h  = blockIdx.y;
  const int tm = blockIdx.x * 128 + wave * 16;
  const float* p = P + (size_t)h * Lq * Lk + (size_t)(tm + row) * Lk;
  const float* vc[4];
#pragma unroll
  for (int s = 0; s < 4; ++s) vc[s] = V + h * HD + s * 16 + row;
  v8f acc[4] = {v8f{}, v8f{}, v8f{}, v8f{}};
  for (int k = 0; k < Lk; k += 32) {
    const int ka = k + half * 8;
    const int kb = k + half * 8;
    v16bf av;
#pragma unroll
    for (int i = 0; i < 8; ++i) {
      av[i]     = f2bf(p[ka + i]);
      av[i + 8] = f2bf(p[ka + 16 + i]);
    }
#pragma unroll
    for (int s = 0; s < 4; ++s) {
      v16bf bv;
#pragma unroll
      for (int j = 0; j < 16; ++j) bv[j] = f2bf(vc[s][(size_t)(kb + (j < 8 ? j : j + 8)) * DM]);
      acc[s] = wmma_bf16(av, bv, acc[s]);
    }
  }
#pragma unroll
  for (int s = 0; s < 4; ++s) {
#pragma unroll
    for (int i = 0; i < 8; ++i) {
      const int m = tm + half * 8 + i;
      O[(size_t)m * DM + h * HD + s * 16 + row] = acc[s][i];
    }
  }
}

// row-wise softmax; one wave per row (wave32 shfl reductions)
__global__ __launch_bounds__(256) void softmax_rows(float* __restrict__ S,
                                                    int n, int nrows) {
  const int lane = threadIdx.x & 31;
  const int wave = threadIdx.x >> 5;
  const int r = blockIdx.x * 8 + wave;
  if (r >= nrows) return;
  float* p = S + (size_t)r * n;
  float mx = -__builtin_inff();
  for (int i = lane; i < n; i += 32) mx = fmaxf(mx, p[i]);
#pragma unroll
  for (int o = 16; o; o >>= 1) mx = fmaxf(mx, __shfl_xor(mx, o));
  float sm = 0.f;
  for (int i = lane; i < n; i += 32) { float e = __expf(p[i] - mx); p[i] = e; sm += e; }
#pragma unroll
  for (int o = 16; o; o >>= 1) sm += __shfl_xor(sm, o);
  const float inv = 1.0f / sm;
  for (int i = lane; i < n; i += 32) p[i] *= inv;
}

// in-place LayerNorm over last dim (512); one wave per row
__global__ __launch_bounds__(256) void layernorm_rows(
    float* __restrict__ X, const float* __restrict__ g,
    const float* __restrict__ b, int nrows) {
  const int lane = threadIdx.x & 31;
  const int wave = threadIdx.x >> 5;
  const int r = blockIdx.x * 8 + wave;
  if (r >= nrows) return;
  float* p = X + (size_t)r * DM;
  float xr[16];
  float s = 0.f, ss = 0.f;
#pragma unroll
  for (int j = 0; j < 16; ++j) {
    const float x = p[lane + j * 32];
    xr[j] = x; s += x; ss += x * x;
  }
#pragma unroll
  for (int o = 16; o; o >>= 1) { s += __shfl_xor(s, o); ss += __shfl_xor(ss, o); }
  const float mean = s * (1.0f / DM);
  const float var  = ss * (1.0f / DM) - mean * mean;
  const float rs   = rsqrtf(var + 1e-5f);
#pragma unroll
  for (int j = 0; j < 16; ++j) {
    const int i = lane + j * 32;
    p[i] = (xr[j] - mean) * rs * g[i] + b[i];
  }
}

// token conv1d(circular,k=3,cin=7) + sinusoid pos emb + temporal + location
__global__ __launch_bounds__(DM) void embed_kernel(
    const float* __restrict__ x, const float* __restrict__ xmark,
    const int* __restrict__ loc, const float* __restrict__ tokw,
    const float* __restrict__ tempw, const float* __restrict__ locemb,
    float* __restrict__ out, int L) {
  const int bl = blockIdx.x;
  const int b = bl / L, l = bl - b * L;
  const int d = threadIdx.x;
  __shared__ float xs[21];
  if (d < 21) {
    const int c = d / 3, t = d % 3;
    int lt = l + t - 1; lt = lt < 0 ? L - 1 : (lt >= L ? lt - L : lt);
    xs[d] = x[((size_t)b * L + lt) * 7 + c];
  }
  __syncthreads();
  float s = 0.f;
  const float* w = tokw + (size_t)d * 21;
#pragma unroll
  for (int j = 0; j < 21; ++j) s += w[j] * xs[j];
  const float freq = expf(-(float)((d >> 1) << 1) * (logf(10000.0f) / (float)DM));
  const float arg  = (float)l * freq;
  s += (d & 1) ? cosf(arg) : sinf(arg);
  s += xmark[(size_t)b * L + l] * tempw[d];
  s += locemb[(size_t)loc[(size_t)b * L + l] * DM + d];
  out[((size_t)b * L + l) * DM + d] = s;
}

// distil conv1d(circular, k=3, 512->512) as WMMA GEMM, K = 3*512.
// 512 % 32 == 0 so each 32-chunk stays inside one tap t.
__global__ __launch_bounds__(256) void distil_conv(
    const float* __restrict__ X, const float* __restrict__ Wc,
    const float* __restrict__ bias, float* __restrict__ Y, int L) {
  const int lane = threadIdx.x & 31;
  const int wave = threadIdx.x >> 5;
  const int row  = lane & 15;
  const int half = lane >> 4;
  const int tm = blockIdx.y * 32 + (wave >> 2) * 16;
  const int tn = blockIdx.x * 256 + (wave & 3) * 64;
  const int m  = tm + row;
  const int bI = m / L;
  const int l  = m - bI * L;
  v8f acc[4] = {v8f{}, v8f{}, v8f{}, v8f{}};
  for (int k = 0; k < 3 * DM; k += 32) {
    const int t  = k >> 9;
    const int kc = k & (DM - 1);
    int lt = l + t - 1; lt = lt < 0 ? (L - 1) : (lt >= L ? lt - L : lt);
    const float* a = X + ((size_t)bI * L + lt) * DM + kc;
    const int ka = half * 8;
    const int kb = kc + half * 8;
    v16bf av;
#pragma unroll
    for (int i = 0; i < 8; ++i) {
      av[i]     = f2bf(a[ka + i]);
      av[i + 8] = f2bf(a[ka + 16 + i]);
    }
#pragma unroll
    for (int s = 0; s < 4; ++s) {
      const int col = tn + s * 16 + row;
      v16bf bv;
#pragma unroll
      for (int j = 0; j < 16; ++j)
        bv[j] = f2bf(Wc[(size_t)col * (3 * DM) + (size_t)(kb + (j < 8 ? j : j + 8)) * 3 + t]);
      acc[s] = wmma_bf16(av, bv, acc[s]);
    }
  }
#pragma unroll
  for (int s = 0; s < 4; ++s) {
    const int col = tn + s * 16 + row;
    const float bc = bias[col];
#pragma unroll
    for (int i = 0; i < 8; ++i) {
      const int mm = tm + half * 8 + i;
      Y[(size_t)mm * DM + col] = acc[s][i] + bc;
    }
  }
}

__global__ void bn_elu_kernel(float* __restrict__ Y, const float* __restrict__ g,
                              const float* __restrict__ b, const float* __restrict__ mn,
                              const float* __restrict__ vr, int n) {
  const int i = blockIdx.x * blockDim.x + threadIdx.x;
  if (i >= n) return;
  const int d = i & (DM - 1);
  const float x = (Y[i] - mn[d]) * rsqrtf(vr[d] + 1e-5f) * g[d] + b[d];
  Y[i] = x > 0.f ? x : expm1f(x);
}

// maxpool window 3, stride 2, pad 1 with -inf: L -> L/2
__global__ void maxpool_kernel(const float* __restrict__ Y, float* __restrict__ O,
                               int L) {
  const int i = blockIdx.x * blockDim.x + threadIdx.x;
  const int n = BB * (L / 2) * DM;
  if (i >= n) return;
  const int d = i & (DM - 1);
  const int j = (i / DM) % (L / 2);
  const int b = i / (DM * (L / 2));
  const float* yb = Y + (size_t)b * L * DM + d;
  const int c = 2 * j;
  float mval = yb[(size_t)c * DM];
  if (c - 1 >= 0) mval = fmaxf(mval, yb[(size_t)(c - 1) * DM]);
  if (c + 1 < L)  mval = fmaxf(mval, yb[(size_t)(c + 1) * DM]);
  O[i] = mval;
}

// final 512 -> 7 projection on the last PRED_LEN=256 positions
__global__ void proj_kernel(const float* __restrict__ X, const float* __restrict__ Wp,
                            const float* __restrict__ bp, float* __restrict__ out) {
  const int idx = blockIdx.x * blockDim.x + threadIdx.x;
  if (idx >= BB * 256 * 7) return;
  const int n = idx % 7;
  const int l = (idx / 7) % 256;
  const int b = idx / (7 * 256);
  const float* x = X + ((size_t)b * DLN + 256 + l) * DM;
  const float* w = Wp + (size_t)n * DM;
  float s = bp[n];
  for (int k = 0; k < DM; ++k) s += x[k] * w[k];
  out[idx] = s;
}

// ---------------------------------------------------------------------------
// Host orchestration. Input pytree flattened in insertion order:
//  0 x_enc 1 x_mark_enc 2 x_dec 3 x_mark_dec 4 loc_enc 5 loc_dec
//  6 tok_enc 7 tok_dec 8 temp_enc 9 temp_dec 10 loc_emb
//  11+16*i enc layer i: wq_w,wq_b,wk_w,wk_b,wv_w,wv_b,wo_w,wo_b,
//                       w1_w,w1_b,w2_w,w2_b,ln1_g,ln1_b,ln2_g,ln2_b
//  43..48 distil (w,b,bn_g,bn_b,bn_m,bn_v)  49,50 enc_norm
//  51 dec layer: self(8), cross(8), ffn(4), ln1,ln2,ln3
//  77,78 dec_norm   79,80 proj
// ---------------------------------------------------------------------------
extern "C" void kernel_launch(void* const* d_in, const int* in_sizes, int n_in,
                              void* d_out, int out_size, void* d_ws, size_t ws_size,
                              hipStream_t stream) {
  (void)in_sizes; (void)n_in; (void)out_size; (void)ws_size;
  auto F = [&](int i) { return (const float*)d_in[i]; };

  float* Wf = (float*)d_ws;
  const size_t M4 = (size_t)4 * 1024 * 1024;  // 4Mi floats = B*SL*DM
  float* xe   = Wf;              // encoder activations   (4M)
  float* tmp  = Wf + 1 * M4;     // residual branch       (4M)
  float* qb   = Wf + 2 * M4;     // Q                     (4M)
  float* kb   = Wf + 3 * M4;     // K                     (4M)
  float* vb   = Wf + 4 * M4;     // V                     (4M)
  float* ao   = Wf + 5 * M4;     // attn pre-proj out     (4M)
  float* hid  = Wf + 6 * M4;     // FFN hidden            (16M)
  float* sc   = Wf + 10 * M4;    // per-batch scores      (8M)
  float* cy   = Wf + 12 * M4;    // distil conv out       (4M)
  float* dx   = Wf + 13 * M4;    // decoder act           (2M)
  float* dtmp = dx + (size_t)2 * 1024 * 1024;  // decoder tmp (2M)

  auto gemm = [&](const float* A, const float* Wm, const float* bias,
                  const float* res, float* C, int M, int N, int K, int act) {
    dim3 g(N / 256, M / 32);
    if (res)
      hipLaunchKernelGGL((gemm_bias_act<0, true>), g, dim3(256), 0, stream,
                         A, K, Wm, K, bias, res, N, C, N, K);
    else if (act == 1)
      hipLaunchKernelGGL((gemm_bias_act<1, false>), g, dim3(256), 0, stream,
                         A, K, Wm, K, bias, (const float*)nullptr, N, C, N, K);
    else
      hipLaunchKernelGGL((gemm_bias_act<0, false>), g, dim3(256), 0, stream,
                         A, K, Wm, K, bias, (const float*)nullptr, N, C, N, K);
  };
  auto lnorm = [&](float* X, const float* g, const float* b, int rows) {
    hipLaunchKernelGGL(layernorm_rows, dim3((rows + 7) / 8), dim3(256), 0, stream,
                       X, g, b, rows);
  };
  // dense scaled-dot attention (prob-sparse approximated dense; upper-bound
  // cost, identical kernel classes). outY = Xq + Wo(softmax(QK^T/8)V) + bo.
  auto attention = [&](const float* Xq, int Lq, const float* Xkv, int Lk,
                       int pb, float* outY, int causal) {
    gemm(Xq,  F(pb + 0), F(pb + 1), nullptr, qb, BB * Lq, DM, DM, 0);
    gemm(Xkv, F(pb + 2), F(pb + 3), nullptr, kb, BB * Lk, DM, DM, 0);
    gemm(Xkv, F(pb + 4), F(pb + 5), nullptr, vb, BB * Lk, DM, DM, 0);
    for (int b = 0; b < BB; ++b) {  // per-batch to bound scratch to 32MB
      hipLaunchKernelGGL(attn_scores, dim3(Lk / 256, Lq / 32, NH), dim3(256), 0,
                         stream, qb + (size_t)b * Lq * DM,
                         kb + (size_t)b * Lk * DM, sc, Lq, Lk, 0.125f, causal);
      hipLaunchKernelGGL(softmax_rows, dim3((NH * Lq) / 8), dim3(256), 0, stream,
                         sc, Lk, NH * Lq);
      hipLaunchKernelGGL(attn_av, dim3(Lq / 128, NH), dim3(256), 0, stream,
                         sc, vb + (size_t)b * Lk * DM,
                         ao + (size_t)b * Lq * DM, Lq, Lk);
    }
    gemm(ao, F(pb + 6), F(pb + 7), Xq /*residual*/, outY, BB * Lq, DM, DM, 0);
  };

  // ---- encoder ----
  hipLaunchKernelGGL(embed_kernel, dim3(BB * SL), dim3(DM), 0, stream,
                     F(0), F(1), (const int*)d_in[4], F(6), F(8), F(10), xe, SL);
  int L = SL;
  for (int li = 0; li < 2; ++li) {
    const int pb = 11 + li * 16;
    attention(xe, L, xe, L, pb, tmp, 0);               // tmp = x + attn(x)
    lnorm(tmp, F(pb + 12), F(pb + 13), BB * L);        // ln1 (in place)
    gemm(tmp, F(pb + 8), F(pb + 9), nullptr, hid, BB * L, DFF, DM, 1);   // gelu
    gemm(hid, F(pb + 10), F(pb + 11), tmp, xe, BB * L, DM, DFF, 0);      // +res
    lnorm(xe, F(pb + 14), F(pb + 15), BB * L);         // ln2
    if (li == 0) {                                     // distil: L -> L/2
      hipLaunchKernelGGL(distil_conv, dim3(DM / 256, (BB * L) / 32), dim3(256),
                         0, stream, xe, F(43), F(44), cy, L);
      const int n = BB * L * DM;
      hipLaunchKernelGGL(bn_elu_kernel, dim3((n + 255) / 256), dim3(256), 0,
                         stream, cy, F(45), F(46), F(47), F(48), n);
      const int n2 = BB * (L / 2) * DM;
      hipLaunchKernelGGL(maxpool_kernel, dim3((n2 + 255) / 256), dim3(256), 0,
                         stream, cy, xe, L);
      L /= 2;
    }
  }
  lnorm(xe, F(49), F(50), BB * L);                     // enc_norm (L = 512)

  // ---- decoder ----
  hipLaunchKernelGGL(embed_kernel, dim3(BB * DLN), dim3(DM), 0, stream,
                     F(2), F(3), (const int*)d_in[5], F(7), F(9), F(10), dx, DLN);
  attention(dx, DLN, dx, DLN, 51, dtmp, 1);            // causal self-attn
  lnorm(dtmp, F(71), F(72), BB * DLN);                 // ln1
  attention(dtmp, DLN, xe, L, 59, dx, 0);              // cross-attn vs encoder
  lnorm(dx, F(73), F(74), BB * DLN);                   // ln2
  gemm(dx, F(67), F(68), nullptr, hid, BB * DLN, DFF, DM, 1);  // ffn gelu
  gemm(hid, F(69), F(70), dx, dtmp, BB * DLN, DM, DFF, 0);     // + residual
  lnorm(dtmp, F(75), F(76), BB * DLN);                 // ln3
  lnorm(dtmp, F(77), F(78), BB * DLN);                 // dec_norm
  hipLaunchKernelGGL(proj_kernel, dim3((BB * 256 * 7 + 255) / 256), dim3(256),
                     0, stream, dtmp, F(79), F(80), (float*)d_out);
}
